// TemporalAttention_65154653880378
// MI455X (gfx1250) — matrix operand order repaired and checked by reading, not compiled, not run on hardware
//
#include <hip/hip_runtime.h>
#include <hip/hip_bf16.h>

typedef __attribute__((ext_vector_type(16))) _Float16 v16h;
typedef __attribute__((ext_vector_type(4)))  _Float16 v4h;
typedef __attribute__((ext_vector_type(8)))  float    v8f;

__device__ __forceinline__ v8f wmma16(v16h a, v16h b, v8f c) {
  // D = A(16x32 f16) * B(32x16 f16) + C(16x16 f32)
  return __builtin_amdgcn_wmma_f32_16x16x32_f16(false, a, false, b, (short)0, c, false, false);
}

// ---- prep: convert + pack weights into fragment-major f16 layout ----------
// For tile (ks, nt): 32 lanes x 16 slots contiguous per lane:
// slot s of lane l = W[(ks*32 + ((l<16)?0:8) + (s&7) + 16*(s>>3))*ldw + nt*16 + (l&15)]
__global__ void pack_weights_f16(const float* __restrict__ Wqkv,
                                 const float* __restrict__ Wproj,
                                 _Float16* __restrict__ wqp,
                                 _Float16* __restrict__ wpp) {
  const int i = blockIdx.x * blockDim.x + threadIdx.x;
  if (i < 256 * 768) {
    const int s = i & 15, l = (i >> 4) & 31, tile = i >> 9;
    const int nt = tile % 48, ks = tile / 48;
    const int ko = (l < 16) ? 0 : 8;
    wqp[i] = (_Float16)Wqkv[(ks * 32 + ko + (s & 7) + ((s >> 3) << 4)) * 768 + nt * 16 + (l & 15)];
  }
  if (i < 256 * 256) {
    const int s = i & 15, l = (i >> 4) & 31, tile = i >> 9;
    const int nt = tile % 16, ks = tile / 16;
    const int ko = (l < 16) ? 0 : 8;
    wpp[i] = (_Float16)Wproj[(ks * 32 + ko + (s & 7) + ((s >> 3) << 4)) * 256 + nt * 16 + (l & 15)];
  }
}

__global__ __launch_bounds__(256)
void ta_fused_kernel(const float* __restrict__ x, const float* __restrict__ pos,
                     const _Float16* __restrict__ wqp, const _Float16* __restrict__ wpp,
                     const float* __restrict__ bproj, float* __restrict__ out) {
  constexpr int T = 24, C = 256;
  // All buffers hold fragment-packed f16 tiles: [tile][lane(32)][slot(16)]
  __shared__ _Float16 s_xp[16 * 32 * 16];  // x as A-frags (mt*8+ks); reused as probs in st.3
  __shared__ _Float16 s_qp[16 * 32 * 16];  // q as A-frags (mt*8+ks); attn-out overwrites
  __shared__ _Float16 s_kp[16 * 32 * 16];  // k^T as B-frags (h*2+rowtile)
  __shared__ _Float16 s_vp[16 * 32 * 16];  // v as B-frags (h*2+coltile)

  const int tid  = threadIdx.x;
  const int lane = tid & 31;
  const int w    = tid >> 5;                 // 8 waves (wave32)
  const int n15  = lane & 15;
  const int bs   = blockIdx.x;               // one workgroup per (b,s)
  const float* xg = x + (size_t)bs * T * C;

  _Float16* xp_l = s_xp + lane * 16;         // per-lane bases: loads become base+imm
  _Float16* qp_l = s_qp + lane * 16;
  _Float16* kp_l = s_kp + lane * 16;
  _Float16* vp_l = s_vp + lane * 16;
  const _Float16* wq_l = wqp + lane * 16;
  const _Float16* wp_l = wpp + lane * 16;

  // ---- stage 1: (x+pos) -> f16 A-frag layout. (t,c): mt=t>>4, ks=c>>5;
  // lane_d=(t&15)+16*bit3(c); slot0=8*bit4(c)+(c&7): float4 -> 4 slots, 8B aligned.
  for (int i = tid; i < 32 * 64; i += 256) {
    const int t = i >> 6, c4 = (i & 63) << 2;
    float4 v = make_float4(0.f, 0.f, 0.f, 0.f);
    if (t < T) {
      const float4 xv = *(const float4*)(xg + t * C + c4);
      const float4 pv = *(const float4*)(pos + t * C + c4);
      v = make_float4(xv.x + pv.x, xv.y + pv.y, xv.z + pv.z, xv.w + pv.w);
    }
    const int mt = t >> 4, ks = c4 >> 5;
    const int lane_d = (t & 15) + 16 * ((c4 >> 3) & 1);
    const int slot0  = 8 * ((c4 >> 4) & 1) + (c4 & 7);
    v4h h4; h4[0] = (_Float16)v.x; h4[1] = (_Float16)v.y; h4[2] = (_Float16)v.z; h4[3] = (_Float16)v.w;
    *(v4h*)(&s_xp[((mt * 8 + ks) * 32 + lane_d) * 16 + slot0]) = h4;
  }
  __syncthreads();

  // ---- stage 2: qkv = x(32x256) @ Wqkv(256x768); store straight into packed
  // ---- q/k/v fragment buffers (closed-form C->A/B layout maps) -------------
  {
    v8f acc[2][6];
#pragma unroll
    for (int mt = 0; mt < 2; ++mt)
#pragma unroll
      for (int j = 0; j < 6; ++j) acc[mt][j] = (v8f)0.0f;

#pragma unroll
    for (int ks = 0; ks < 8; ++ks) {
      const v16h a0 = *(const v16h*)(xp_l + (0 * 8 + ks) * 512);
      const v16h a1 = *(const v16h*)(xp_l + (1 * 8 + ks) * 512);
#pragma unroll
      for (int j = 0; j < 6; ++j) {
        const v16h b = *(const v16h*)(wq_l + (ks * 48 + w * 6 + j) * 512);
        acc[0][j] = wmma16(a0, b, acc[0][j]);
        acc[1][j] = wmma16(a1, b, acc[1][j]);
      }
    }
#pragma unroll
    for (int mt = 0; mt < 2; ++mt)
#pragma unroll
      for (int j = 0; j < 6; ++j) {
        const int nt = w * 6 + j;                       // wave-uniform
        if (nt < 16) {                                  // ---- q -> A-frag pack
          const int ks = nt >> 1;
          _Float16* base = s_qp + ((mt * 8 + ks) * 32 + 16 * ((n15 >> 3) & 1)) * 16
                         + (n15 & 7) + 8 * (nt & 1);
#pragma unroll
          for (int r = 0; r < 8; ++r) {
            const int m15 = (lane < 16) ? r : r + 8;
            base[m15 * 16] = (_Float16)acc[mt][j][r];
          }
        } else if (nt < 32) {                           // ---- k -> B^T-frag pack
          const int h = (nt - 16) >> 1;
          _Float16* base = s_kp + ((h * 2 + mt) * 32 + 16 * ((n15 >> 3) & 1)) * 16 + (n15 & 7) + 8 * (nt & 1);
#pragma unroll
          for (int r = 0; r < 8; ++r) {
            const int m15 = (lane < 16) ? r : r + 8;
            base[m15 * 16] = (_Float16)acc[mt][j][r];
          }
        } else {                                        // ---- v -> B-frag pack
          const int h = (nt - 32) >> 1;
          _Float16* base = s_vp + ((h * 2 + (nt & 1)) * 32 + n15) * 16 + 8 * mt;
#pragma unroll
          for (int r = 0; r < 8; ++r) {
            const int m15 = (lane < 16) ? r : r + 8;
            base[((m15 >> 3) & 1) * 256 + (m15 & 7)] = (_Float16)acc[mt][j][r];
          }
        }
      }
  }
  __syncthreads();

  // ---- stage 3: one wave per head; all operands are contiguous frag loads ---
  {
    const int h = w;
    _Float16* probsp = s_xp + h * 1024;     // packed probs A-frags [(mt*32+lane)*16+slot]

    // scores = q @ k^T (2x2 tiles, single K=32 step)
    v8f sc[2][2];
    {
      const v16h b0 = *(const v16h*)(kp_l + (h * 2 + 0) * 512);
      const v16h b1 = *(const v16h*)(kp_l + (h * 2 + 1) * 512);
#pragma unroll
      for (int mt = 0; mt < 2; ++mt) {
        sc[mt][0] = (v8f)0.0f; sc[mt][1] = (v8f)0.0f;
        const v16h a = *(const v16h*)(qp_l + (mt * 8 + h) * 512);
        sc[mt][0] = wmma16(a, b0, sc[mt][0]);
        sc[mt][1] = wmma16(a, b1, sc[mt][1]);
      }
    }

    // causal mask + row softmax; write probs in packed A-frag layout:
    // lane_d=(M&15)+16*bit3(N), slot=(N&7)+8*(N>>4)
    const float scale = 0.17677669529663687f;           // 1/sqrt(32)
#pragma unroll
    for (int mt = 0; mt < 2; ++mt)
#pragma unroll
      for (int r = 0; r < 8; ++r) {
        const int M  = mt * 16 + ((lane < 16) ? r : r + 8);
        const int N0 = n15, N1 = 16 + n15;
        float s0 = sc[mt][0][r] * scale;
        float s1 = sc[mt][1][r] * scale;
        if (N0 > M) s0 = -1e30f;
        if (N1 > M) s1 = -1e30f;
        float mx = fmaxf(s0, s1);
#pragma unroll
        for (int o = 8; o >= 1; o >>= 1) mx = fmaxf(mx, __shfl_xor(mx, o, 16));
        const float e0 = __expf(s0 - mx), e1 = __expf(s1 - mx);
        float sum = e0 + e1;
#pragma unroll
        for (int o = 8; o >= 1; o >>= 1) sum += __shfl_xor(sum, o, 16);
        const float inv = 1.0f / sum;
        const int m15 = M & 15;
        const int ld0 = m15 + 16 * ((N0 >> 3) & 1);
        probsp[(mt * 32 + ld0) * 16 + (N0 & 7)] = (_Float16)(e0 * inv);
        const int ld1 = m15 + 16 * ((N1 >> 3) & 1);
        probsp[(mt * 32 + ld1) * 16 + (N1 & 7) + 8] = (_Float16)(e1 * inv);
      }

    // out_h = P(32x32) @ v(32x32); scatter into s_qp tiles ks=h (A-frag layout
    // for stage 4). Only this wave touches those tiles between barriers.
    v8f oa[2][2];
    {
      const v16h b0 = *(const v16h*)(vp_l + (h * 2 + 0) * 512);
      const v16h b1 = *(const v16h*)(vp_l + (h * 2 + 1) * 512);
#pragma unroll
      for (int mt = 0; mt < 2; ++mt) {
        oa[mt][0] = (v8f)0.0f; oa[mt][1] = (v8f)0.0f;
        const v16h a = *(const v16h*)(probsp + (mt * 32 + lane) * 16);
        oa[mt][0] = wmma16(a, b0, oa[mt][0]);
        oa[mt][1] = wmma16(a, b1, oa[mt][1]);
      }
    }
#pragma unroll
    for (int mt = 0; mt < 2; ++mt)
#pragma unroll
      for (int nt = 0; nt < 2; ++nt) {
        _Float16* base = s_qp + ((mt * 8 + h) * 32 + 16 * ((n15 >> 3) & 1)) * 16
                       + (n15 & 7) + 8 * nt;
#pragma unroll
        for (int r = 0; r < 8; ++r) {
          const int m15 = (lane < 16) ? r : r + 8;
          base[m15 * 16] = (_Float16)oa[mt][nt][r];
        }
      }
  }
  __syncthreads();

  // ---- stage 4: out = attn(32x256) @ Wproj(256x256) + b ---------------------
  {
    v8f acc[2][2];
#pragma unroll
    for (int mt = 0; mt < 2; ++mt) { acc[mt][0] = (v8f)0.0f; acc[mt][1] = (v8f)0.0f; }

#pragma unroll
    for (int ks = 0; ks < 8; ++ks) {
      const v16h a0 = *(const v16h*)(qp_l + (0 * 8 + ks) * 512);
      const v16h a1 = *(const v16h*)(qp_l + (1 * 8 + ks) * 512);
#pragma unroll
      for (int j = 0; j < 2; ++j) {
        const v16h b = *(const v16h*)(wp_l + (ks * 16 + w * 2 + j) * 512);
        acc[0][j] = wmma16(a0, b, acc[0][j]);
        acc[1][j] = wmma16(a1, b, acc[1][j]);
      }
    }

    float* ob = out + (size_t)bs * T * C;
#pragma unroll
    for (int j = 0; j < 2; ++j) {
      const int n = (w * 2 + j) * 16 + n15;
      const float bias = bproj[n];
#pragma unroll
      for (int mt = 0; mt < 2; ++mt)
#pragma unroll
        for (int r = 0; r < 8; ++r) {
          const int m = mt * 16 + ((lane < 16) ? r : r + 8);
          if (m < T) ob[m * C + n] = acc[mt][j][r] + bias;
        }
    }
  }
}

extern "C" void kernel_launch(void* const* d_in, const int* in_sizes, int n_in,
                              void* d_out, int out_size, void* d_ws, size_t ws_size,
                              hipStream_t stream) {
  const float* x     = (const float*)d_in[0];   // (4,1024,24,256)
  const float* pos   = (const float*)d_in[1];   // (24,256)
  const float* Wqkv  = (const float*)d_in[2];   // (256,768)
  const float* Wproj = (const float*)d_in[3];   // (256,256)
  const float* bproj = (const float*)d_in[4];   // (256,)
  float* out = (float*)d_out;

  _Float16* wqp = (_Float16*)d_ws;              // packed Wqkv fragments, 384 KB
  _Float16* wpp = wqp + 256 * 768;              // packed Wproj fragments, 128 KB

  pack_weights_f16<<<(256 * 768 + 255) / 256, 256, 0, stream>>>(Wqkv, Wproj, wqp, wpp);

  const int blocks = 4 * 1024;                  // one workgroup per (b,s)
  ta_fused_kernel<<<blocks, 256, 0, stream>>>(x, pos, wqp, wpp, bproj, out);
}
